// Resblock_59725815218374
// MI455X (gfx1250) — matrix operand order repaired and checked by reading, not compiled, not run on hardware
//
#include <hip/hip_runtime.h>

typedef __attribute__((ext_vector_type(16))) _Float16 v16h;
typedef __attribute__((ext_vector_type(8)))  float    v8f;

#define CIN   128
#define HH    64
#define WW    384
#define BB    16
#define TILE_W 64
#define LPITCH 136   // halfs per (row,px) LDS entry: 128 + 8 pad (bank-conflict avoidance)
#define EPSV  1e-5f

struct alignas(16) H16 { uint4 lo, hi; };
struct alignas(16) H8  { _Float16 h[8]; };

__device__ inline v16h load_v16h(const _Float16* p) {
    H16 r = *(const H16*)p;              // two 16B loads (16B-aligned is enough)
    return __builtin_bit_cast(v16h, r);
}

// ---------------------------------------------------------------------------
// Fold BN/bias/softmax into per-channel affine terms.
// sf = softmax(W); conv1: y*a1+b1 then relu. conv2: y*a2+b2 (sf0 folded in).
// ---------------------------------------------------------------------------
__global__ void prep_scalars(const float* __restrict__ Wv,
                             const float* __restrict__ c1b,
                             const float* __restrict__ g1, const float* __restrict__ b1,
                             const float* __restrict__ m1, const float* __restrict__ v1,
                             const float* __restrict__ c2b,
                             const float* __restrict__ g2, const float* __restrict__ b2,
                             const float* __restrict__ m2, const float* __restrict__ v2,
                             float* __restrict__ sfarr,
                             float* __restrict__ a1, float* __restrict__ be1,
                             float* __restrict__ a2, float* __restrict__ be2) {
    int t = threadIdx.x;
    float e0 = __expf(Wv[0]), e1 = __expf(Wv[1]);
    float inv = 1.0f / (e0 + e1);
    float sf0 = e0 * inv, sf1 = e1 * inv;
    if (t == 0) { sfarr[0] = sf0; sfarr[1] = sf1; }
    if (t < 128) {
        float s = g1[t] * rsqrtf(v1[t] + EPSV);
        a1[t]  = s;
        be1[t] = c1b[t] * s + b1[t] - m1[t] * s;
    }
    if (t < 96) {
        float s = g2[t] * rsqrtf(v2[t] + EPSV);
        a2[t]  = s * sf0;
        be2[t] = (c2b[t] * s + b2[t] - m2[t] * s) * sf0;
    }
}

// ---------------------------------------------------------------------------
// BN0 + ReLU + fp32->f16 + NCHW->NHWC via LDS transpose (coalesced both ways).
// One block = 32 pixels of one (b,y) row.
// ---------------------------------------------------------------------------
__global__ __launch_bounds__(256) void prep_input(
        const float* __restrict__ x,
        const float* __restrict__ g, const float* __restrict__ bt,
        const float* __restrict__ m, const float* __restrict__ v,
        _Float16* __restrict__ h0) {
    __shared__ __align__(16) float tile[32 * 132];
    const int tid = threadIdx.x;
    const int x0  = blockIdx.x * 32;
    const int y   = blockIdx.y;
    const int b   = blockIdx.z;
    #pragma unroll
    for (int i = 0; i < 16; ++i) {
        int idx = tid + i * 256;            // 0..4095
        int ci = idx >> 5, px = idx & 31;   // lanes sweep px -> coalesced reads
        float s   = g[ci] * rsqrtf(v[ci] + EPSV);
        float val = x[(((size_t)b * CIN + ci) * HH + y) * WW + x0 + px];
        val = val * s + (bt[ci] - m[ci] * s);
        tile[px * 132 + ci] = val > 0.f ? val : 0.f;
    }
    __syncthreads();
    #pragma unroll
    for (int i = 0; i < 2; ++i) {
        int idx = tid + i * 256;            // 0..511
        int px = idx >> 4, cq = idx & 15;   // lanes sweep ci-octets -> coalesced writes
        H8 o;
        #pragma unroll
        for (int r = 0; r < 8; ++r) o.h[r] = (_Float16)tile[px * 132 + cq * 8 + r];
        *(H8*)(h0 + (((size_t)b * HH + y) * WW + x0 + px) * CIN + cq * 8) = o;
    }
}

// ---------------------------------------------------------------------------
// Repack OIHW fp32 weights into the per-lane WMMA A-fragment layout (f16):
// frag[(((tap*4+kc)*COT+cot)*32+lane)*16 + j], lane<16: K={0..15},
// lane>=16: K={16..31}  (contiguous K order, matching the B fragment fill).
// ---------------------------------------------------------------------------
__global__ void repack_w(const float* __restrict__ w, _Float16* __restrict__ frag,
                         int COUT) {
    int idx = blockIdx.x * 256 + threadIdx.x;
    int cot_n = COUT >> 4;
    int total = 9 * 4 * cot_n * 32 * 16;
    if (idx >= total) return;
    int j    = idx & 15;
    int lane = (idx >> 4) & 31;
    int cot  = (idx >> 9) % cot_n;
    int rest = idx / (512 * cot_n);
    int kc   = rest & 3;
    int tap  = rest >> 2;
    int co = cot * 16 + (lane & 15);
    int K  = j + (lane & 16);
    int ci = kc * 32 + K;
    int ky = tap / 3, kx = tap - ky * 3;
    frag[idx] = (_Float16)w[((co * CIN + ci) * 3 + ky) * 3 + kx];
}

// ---------------------------------------------------------------------------
// Implicit-GEMM 3x3 conv with v_wmma_f32_16x16x32_f16.
// Block = (b, y, 64-px strip). 8 waves; wave w owns co-tile [16w,16w+16).
// LDS holds 3x66x128 f16 input slab (NHWC, pitch 136 -> conflict-free b128 DS).
// FINAL=false: +affine+relu -> f16 NHWC out (via LDS transpose store).
// FINAL=true : +affine(sf0 folded) + packed-channel skip*sf1 -> fp32 NCHW out.
// ---------------------------------------------------------------------------
template <int COUT, bool FINAL>
__global__ __launch_bounds__(256) void conv3x3_wmma(
        const _Float16* __restrict__ in,     // [B,H,W,128] f16
        const _Float16* __restrict__ wfrag,  // fragment-ordered weights
        const float* __restrict__ alpha, const float* __restrict__ beta,
        const float* __restrict__ sfarr,     // FINAL only
        const float* __restrict__ xorig,     // FINAL only: x NCHW fp32
        void* __restrict__ outp) {
    constexpr int KC  = CIN / 32;   // 4
    constexpr int COT = COUT / 16;  // 8 or 6
    __shared__ __align__(16) _Float16 lds[3 * 66 * LPITCH];  // 53,856 B

    const int tid  = threadIdx.x;
    const int wave = tid >> 5;
    const int lane = tid & 31;
    const int x0   = blockIdx.x * TILE_W;
    const int y    = blockIdx.y;
    const int b    = blockIdx.z;

    // ---- stage input slab: 3 rows x 66 px x 128 ci, zero-padded halo ----
    for (int idx = tid; idx < 3 * 66 * 16; idx += 256) {
        int row = idx / (66 * 16);
        int rem = idx - row * (66 * 16);
        int px = rem >> 4, cq = rem & 15;
        int yy = y + row - 1;
        int xx = x0 + px - 1;
        uint4 val = {0u, 0u, 0u, 0u};
        if (yy >= 0 && yy < HH && xx >= 0 && xx < WW)
            val = *(const uint4*)(in + (((size_t)b * HH + yy) * WW + xx) * CIN + cq * 8);
        *(uint4*)(&lds[(row * 66 + px) * LPITCH + cq * 8]) = val;
    }
    __syncthreads();

    const int lane16 = lane & 15;
    const int koff   = (lane & 16);  // 0 or 16: K offset of this half-wave's B frag

    v8f acc[4];
    #pragma unroll
    for (int t = 0; t < 4; ++t)
        #pragma unroll
        for (int e = 0; e < 8; ++e) acc[t][e] = 0.f;

    if (wave < COT) {
        for (int tap = 0; tap < 9; ++tap) {
            const int ky = tap / 3, kx = tap - ky * 3;
            const _Float16* lrow = &lds[(ky * 66 + kx) * LPITCH + koff];
            #pragma unroll
            for (int kc = 0; kc < KC; ++kc) {
                v16h a = load_v16h(
                    wfrag + ((((size_t)tap * KC + kc) * COT + wave) * 32 + lane) * 16);
                #pragma unroll
                for (int t = 0; t < 4; ++t) {
                    v16h bf = load_v16h(lrow + (t * 16 + lane16) * LPITCH + kc * 32);
                    acc[t] = __builtin_amdgcn_wmma_f32_16x16x32_f16(
                        false, a, false, bf, (short)0, acc[t], false, false);
                }
            }
        }
    }
    __syncthreads();  // everyone done reading the input slab

    const int cobase = wave * 16 + ((lane & 16) >> 1);  // +8 for upper half-wave

    if constexpr (!FINAL) {
        // affine + relu -> f16, transpose through LDS for coalesced NHWC store
        if (wave < COT) {
            #pragma unroll
            for (int t = 0; t < 4; ++t) {
                int px = t * 16 + lane16;
                #pragma unroll
                for (int r = 0; r < 8; ++r) {
                    int co = cobase + r;
                    float v = acc[t][r] * alpha[co] + beta[co];
                    v = v > 0.f ? v : 0.f;
                    lds[px * LPITCH + co] = (_Float16)v;
                }
            }
        }
        __syncthreads();
        _Float16* out = (_Float16*)outp;
        for (int idx = tid; idx < TILE_W * 16; idx += 256) {
            int px = idx >> 4, cq = idx & 15;
            uint4 val = *(const uint4*)(&lds[px * LPITCH + cq * 8]);
            *(uint4*)(out + (((size_t)b * HH + y) * WW + x0 + px) * CIN + cq * 8) = val;
        }
    } else {
        // affine (sf0 folded) -> fp32 staging, then + skip*sf1, NCHW store
        float* ldsf = (float*)lds;  // [COUT][TILE_W] floats = 24,576 B
        if (wave < COT) {
            #pragma unroll
            for (int t = 0; t < 4; ++t) {
                int px = t * 16 + lane16;
                #pragma unroll
                for (int r = 0; r < 8; ++r) {
                    int co = cobase + r;
                    ldsf[co * TILE_W + px] = acc[t][r] * alpha[co] + beta[co];
                }
            }
        }
        __syncthreads();
        const float sf1 = sfarr[1];
        float* out = (float*)outp;
        for (int idx = tid; idx < COUT * TILE_W; idx += 256) {
            int co = idx / TILE_W, px = idx - co * TILE_W;
            int cin_skip = ((co / 3) << 2) + (co - (co / 3) * 3);  // IDX[co]=4*(co/3)+co%3
            float skip = xorig[(((size_t)b * CIN + cin_skip) * HH + y) * WW + x0 + px] * sf1;
            out[(((size_t)b * COUT + co) * HH + y) * WW + x0 + px] = ldsf[idx] + skip;
        }
    }
}

// ---------------------------------------------------------------------------
extern "C" void kernel_launch(void* const* d_in, const int* in_sizes, int n_in,
                              void* d_out, int out_size, void* d_ws, size_t ws_size,
                              hipStream_t stream) {
    (void)in_sizes; (void)n_in; (void)out_size; (void)ws_size;
    const float* x    = (const float*)d_in[0];
    const float* Wv   = (const float*)d_in[1];
    const float* g0   = (const float*)d_in[2];
    const float* b0   = (const float*)d_in[3];
    const float* m0   = (const float*)d_in[4];
    const float* v0   = (const float*)d_in[5];
    const float* c1w  = (const float*)d_in[6];
    const float* c1b  = (const float*)d_in[7];
    const float* g1   = (const float*)d_in[8];
    const float* b1   = (const float*)d_in[9];
    const float* m1   = (const float*)d_in[10];
    const float* v1   = (const float*)d_in[11];
    const float* c2w  = (const float*)d_in[12];
    const float* c2b  = (const float*)d_in[13];
    const float* g2   = (const float*)d_in[14];
    const float* b2   = (const float*)d_in[15];
    const float* m2   = (const float*)d_in[16];
    const float* v2   = (const float*)d_in[17];

    char* ws = (char*)d_ws;
    float* sfarr = (float*)ws;          // 2 floats
    float* a1  = sfarr + 16;            // 128
    float* be1 = a1 + 128;              // 128
    float* a2  = be1 + 128;             // 96
    float* be2 = a2 + 96;               // 96
    _Float16* wf1 = (_Float16*)(ws + 4096);            // 9*4*8*512  = 147456 halfs
    _Float16* wf2 = (_Float16*)(ws + 4096 + 294912);   // 9*4*6*512  = 110592 halfs
    const size_t h_elems = (size_t)BB * HH * WW * CIN; // 50,331,648 halfs each
    _Float16* h0 = (_Float16*)(ws + (1ull << 20));
    _Float16* h1 = (_Float16*)(ws + (1ull << 20) + h_elems * 2);

    prep_scalars<<<1, 128, 0, stream>>>(Wv, c1b, g1, b1, m1, v1,
                                        c2b, g2, b2, m2, v2,
                                        sfarr, a1, be1, a2, be2);
    prep_input<<<dim3(WW / 32, HH, BB), 256, 0, stream>>>(x, g0, b0, m0, v0, h0);
    repack_w<<<(9 * 4 * 8 * 512 + 255) / 256, 256, 0, stream>>>(c1w, wf1, 128);
    repack_w<<<(9 * 4 * 6 * 512 + 255) / 256, 256, 0, stream>>>(c2w, wf2, 96);

    conv3x3_wmma<128, false><<<dim3(WW / TILE_W, HH, BB), 256, 0, stream>>>(
        h0, wf1, a1, be1, nullptr, nullptr, (void*)h1);
    conv3x3_wmma<96, true><<<dim3(WW / TILE_W, HH, BB), 256, 0, stream>>>(
        h1, wf2, a2, be2, sfarr, x, d_out);
}
